// Interfuser_83683142795414
// MI455X (gfx1250) — matrix operand order repaired and checked by reading, not compiled, not run on hardware
//
#include <hip/hip_runtime.h>

#define L 151
#define D 256
#define H 8
#define DH 32
#define FF 2048
#define NL 6
#define BATCH 256
#define M_TOT (BATCH * L)        // 38656 = 128 * 302
#define MT_TOT 302               // M tiles of 128
#define CHT 38                   // M tiles per FF chunk

typedef __bf16 bf16x8  __attribute__((ext_vector_type(8)));
typedef __bf16 bf16x16 __attribute__((ext_vector_type(16)));
typedef float  f32x8   __attribute__((ext_vector_type(8)));

__device__ __forceinline__ __bf16 f2bf(float f) {
    unsigned u = __builtin_bit_cast(unsigned, f);
    u += 0x7fffu + ((u >> 16) & 1u);
    unsigned short hs = (unsigned short)(u >> 16);
    return __builtin_bit_cast(__bf16, hs);
}

__device__ __forceinline__ bf16x16 bcat(bf16x8 lo, bf16x8 hi) {
    bf16x16 r;
#pragma unroll
    for (int j = 0; j < 8; ++j) { r[j] = lo[j]; r[8 + j] = hi[j]; }
    return r;
}

__device__ __forceinline__ int blkid(int p) {
    return (p < 50) ? 0 : (p < 67) ? 1 : (p < 84) ? 2 : (p < 101) ? 3 : 4;
}

// ---------------- f32 -> bf16 convert ----------------
__global__ void k_cvt(const float* __restrict__ in, __bf16* __restrict__ out, int n) {
    int i = (blockIdx.x * blockDim.x + threadIdx.x) * 4;
#pragma unroll
    for (int j = 0; j < 4; ++j)
        if (i + j < n) out[i + j] = f2bf(in[i + j]);
}

// ---------------- WMMA GEMM: C[M,N] = A[M,K] @ W[N,K]^T + bias ----------------
// WG tile 128x128, 8 waves (4 in M x 2 in N), wave tile 32x64 = 2x4 WMMA frags.
// Double-buffered LDS, one barrier per K-step. All shapes compile-time.
template <int N, int K, bool RELU, bool WF32, bool WBF16>
__global__ __launch_bounds__(256) void k_gemm(
    const __bf16* __restrict__ A, const __bf16* __restrict__ W,
    const float* __restrict__ bias, float* __restrict__ Cf,
    __bf16* __restrict__ Cb)
{
    __shared__ __attribute__((aligned(16))) __bf16 As[2][128 * 32];
    __shared__ __attribute__((aligned(16))) __bf16 Ws[2][128 * 32];
    const int tid  = threadIdx.x;
    const int lane = tid & 31;
    const int wid  = tid >> 5;
    const int wm   = wid & 3;          // 0..3 -> M rows wm*32
    const int wn   = wid >> 2;         // 0..1 -> N cols wn*64
    const int c    = lane & 15;
    const int h    = lane >> 4;
    const size_t m0 = (size_t)blockIdx.y * 128;
    const int    n0 = blockIdx.x * 128;

    // cooperative tile loads: 128x32 each for A and W, 16 elems/thread each
    const int trow = tid >> 1, tkk = (tid & 1) * 16;
    const __bf16* gA = A + (m0 + trow) * (size_t)K + tkk;
    const __bf16* gW = W + (size_t)(n0 + trow) * K + tkk;

    f32x8 acc[2][4] = {};
    bf16x8 pa0, pa1, pw0, pw1;

    // prologue: fetch k-slice 0
    pa0 = ((const bf16x8*)gA)[0]; pa1 = ((const bf16x8*)gA)[1];
    pw0 = ((const bf16x8*)gW)[0]; pw1 = ((const bf16x8*)gW)[1];
    {
        __bf16* as = &As[0][trow * 32 + tkk];
        *(bf16x8*)as = pa0; *(bf16x8*)(as + 8) = pa1;
        __bf16* ws = &Ws[0][trow * 32 + tkk];
        *(bf16x8*)ws = pw0; *(bf16x8*)(ws + 8) = pw1;
    }

    constexpr int NK = K / 32;
    for (int it = 0; it < NK; ++it) {
        __syncthreads();
        const int cur = it & 1;
        if (it + 1 < NK) {
            const __bf16* ga = gA + (it + 1) * 32;
            const __bf16* gw = gW + (it + 1) * 32;
            pa0 = ((const bf16x8*)ga)[0]; pa1 = ((const bf16x8*)ga)[1];
            pw0 = ((const bf16x8*)gw)[0]; pw1 = ((const bf16x8*)gw)[1];
        }
        bf16x16 af[2], bw[4];
#pragma unroll
        for (int i = 0; i < 2; ++i) {
            const __bf16* pr = &As[cur][(wm * 32 + i * 16 + c) * 32];
            af[i] = bcat(*(const bf16x8*)(pr + h * 8), *(const bf16x8*)(pr + 16 + h * 8));
        }
#pragma unroll
        for (int j = 0; j < 4; ++j) {
            const __bf16* pr = &Ws[cur][(wn * 64 + j * 16 + c) * 32];
            bw[j] = bcat(*(const bf16x8*)(pr + h * 8), *(const bf16x8*)(pr + 16 + h * 8));
        }
#pragma unroll
        for (int i = 0; i < 2; ++i)
#pragma unroll
            for (int j = 0; j < 4; ++j)
                acc[i][j] = __builtin_amdgcn_wmma_f32_16x16x32_bf16(
                    false, af[i], false, bw[j], (short)0, acc[i][j], false, false);
        if (it + 1 < NK) {
            const int nxt = (it + 1) & 1;
            __bf16* as = &As[nxt][trow * 32 + tkk];
            *(bf16x8*)as = pa0; *(bf16x8*)(as + 8) = pa1;
            __bf16* ws = &Ws[nxt][trow * 32 + tkk];
            *(bf16x8*)ws = pw0; *(bf16x8*)(ws + 8) = pw1;
        }
    }

    // epilogue: one base pointer; all offsets compile-time immediates
    const size_t base = (m0 + wm * 32 + 8 * h) * (size_t)N + n0 + wn * 64 + c;
    float  bv[4];
#pragma unroll
    for (int j = 0; j < 4; ++j) bv[j] = bias[n0 + wn * 64 + j * 16 + c];
    float*  cf = WF32  ? (Cf + base) : nullptr;
    __bf16* cb = WBF16 ? (Cb + base) : nullptr;
#pragma unroll
    for (int i = 0; i < 2; ++i)
#pragma unroll
        for (int j = 0; j < 4; ++j)
#pragma unroll
            for (int v = 0; v < 8; ++v) {
                float val = acc[i][j][v] + bv[j];
                if (RELU) val = fmaxf(val, 0.f);
                if (WF32)  cf[(size_t)(i * 16 + v) * N + j * 16] = val;
                if (WBF16) cb[(size_t)(i * 16 + v) * N + j * 16] = f2bf(val);
            }
}

// ---------------- attention: one block per (batch, head) ----------------
// block = 160 threads (5 waves); each wave handles 2 query strips of 16 rows.
__global__ __launch_bounds__(160) void k_attn(const __bf16* __restrict__ qkv,
                                              __bf16* __restrict__ ctx)
{
    __shared__ __attribute__((aligned(16))) __bf16 Ks[160 * 32];   // K  [l][d], padded
    __shared__ __attribute__((aligned(16))) __bf16 Vt[32 * 160];   // V^T[d][l], padded
    __shared__ __attribute__((aligned(16))) __bf16 Pb[5 * 16 * 160];
    const int tid = threadIdx.x, nth = 160;
    const int bb = blockIdx.x >> 3, hd = blockIdx.x & 7;
    const int lane = tid & 31, w = tid >> 5;
    const int c = lane & 15, h = lane >> 4;
    const __bf16 z = __builtin_bit_cast(__bf16, (unsigned short)0);

    for (int idx = tid; idx < 288; idx += nth)
        Ks[(151 + (idx >> 5)) * 32 + (idx & 31)] = z;
    for (int idx = tid; idx < 288; idx += nth)
        Vt[(idx / 9) * 160 + 151 + (idx % 9)] = z;
    for (int idx = tid; idx < 151 * 32; idx += nth) {
        int l = idx >> 5, d = idx & 31;
        size_t base = ((size_t)bb * L + l) * 768 + hd * 32 + d;
        Ks[l * 32 + d]  = qkv[base + 256];
        Vt[d * 160 + l] = qkv[base + 512];
    }
    __syncthreads();

    __bf16* P = &Pb[w * 2560];
    const float scale = 0.17677669529663687f;

    for (int s = w; s < 10; s += 5) {
        const int m0 = s * 16;
        int qrow = m0 + c; if (qrow > 150) qrow = 150;
        const __bf16* qp = qkv + ((size_t)bb * L + qrow) * 768 + hd * 32;
        bf16x16 aQ = bcat(*(const bf16x8*)(qp + h * 8), *(const bf16x8*)(qp + 16 + h * 8));

        f32x8 acc[10] = {};
#pragma unroll
        for (int t = 0; t < 10; ++t) {
            const __bf16* kp = &Ks[(t * 16 + c) * 32];
            bf16x16 bK = bcat(*(const bf16x8*)(kp + h * 8), *(const bf16x8*)(kp + 16 + h * 8));
            acc[t] = __builtin_amdgcn_wmma_f32_16x16x32_bf16(
                false, aQ, false, bK, (short)0, acc[t], false, false);
        }

        // mask + softmax in-register (each row lives in one 16-lane half-wave)
#pragma unroll
        for (int v = 0; v < 8; ++v) {
            int qr = m0 + v + 8 * h;
            int qb = blkid(qr);
            float sv[10];
            float mx = -3.0e38f;
#pragma unroll
            for (int t = 0; t < 10; ++t) {
                int kc = t * 16 + c;
                float bias = (kc < 151 && blkid(kc) == qb) ? 0.f : -1e9f;
                sv[t] = acc[t][v] * scale + bias;
                mx = fmaxf(mx, sv[t]);
            }
            mx = fmaxf(mx, __shfl_xor(mx, 1, 32));
            mx = fmaxf(mx, __shfl_xor(mx, 2, 32));
            mx = fmaxf(mx, __shfl_xor(mx, 4, 32));
            mx = fmaxf(mx, __shfl_xor(mx, 8, 32));
            float ssum = 0.f;
#pragma unroll
            for (int t = 0; t < 10; ++t) { sv[t] = __expf(sv[t] - mx); ssum += sv[t]; }
            ssum += __shfl_xor(ssum, 1, 32);
            ssum += __shfl_xor(ssum, 2, 32);
            ssum += __shfl_xor(ssum, 4, 32);
            ssum += __shfl_xor(ssum, 8, 32);
            float inv = 1.f / ssum;
#pragma unroll
            for (int t = 0; t < 10; ++t)
                P[(v + 8 * h) * 160 + t * 16 + c] = f2bf(sv[t] * inv);
        }

        // ctx = P(16x160) @ V(160x32): 5 K-steps x 2 N-tiles
        f32x8 o[2] = {};
#pragma unroll
        for (int kt = 0; kt < 5; ++kt) {
            const __bf16* pr = &P[c * 160 + kt * 32];
            bf16x16 aP = bcat(*(const bf16x8*)(pr + h * 8), *(const bf16x8*)(pr + 16 + h * 8));
#pragma unroll
            for (int n = 0; n < 2; ++n) {
                const __bf16* vp = &Vt[(n * 16 + c) * 160 + kt * 32];
                bf16x16 bV = bcat(*(const bf16x8*)(vp + h * 8), *(const bf16x8*)(vp + 16 + h * 8));
                o[n] = __builtin_amdgcn_wmma_f32_16x16x32_bf16(
                    false, aP, false, bV, (short)0, o[n], false, false);
            }
        }
#pragma unroll
        for (int n = 0; n < 2; ++n)
#pragma unroll
            for (int v = 0; v < 8; ++v) {
                int lr = m0 + v + 8 * h;
                if (lr < 151)
                    ctx[((size_t)bb * L + lr) * 256 + hd * 32 + n * 16 + c] = f2bf(o[n][v]);
            }
    }
}

// ---------------- residual add + LayerNorm (one block per row of 256) ----------------
__global__ __launch_bounds__(256) void k_ln(
    const float* __restrict__ xin, const float* __restrict__ y,
    const float* __restrict__ sc, const float* __restrict__ bi,
    float* __restrict__ xout, __bf16* __restrict__ xb)
{
    __shared__ float red[16];
    const int i = threadIdx.x;
    const size_t idx = (size_t)blockIdx.x * 256 + i;
    float t = xin[idx] + y[idx];
    float s1 = t, s2 = t * t;
#pragma unroll
    for (int m = 16; m; m >>= 1) {
        s1 += __shfl_xor(s1, m, 32);
        s2 += __shfl_xor(s2, m, 32);
    }
    if ((i & 31) == 0) { red[i >> 5] = s1; red[8 + (i >> 5)] = s2; }
    __syncthreads();
    float a = 0.f, b2 = 0.f;
#pragma unroll
    for (int k = 0; k < 8; ++k) { a += red[k]; b2 += red[8 + k]; }
    float mu  = a * (1.f / 256.f);
    float var = b2 * (1.f / 256.f) - mu * mu;
    float inv = rsqrtf(var + 1e-5f);
    float o = (t - mu) * inv * sc[i] + bi[i];
    xout[idx] = o;
    xb[idx] = f2bf(o);
}

extern "C" void kernel_launch(void* const* d_in, const int* in_sizes, int n_in,
                              void* d_out, int out_size, void* d_ws, size_t ws_size,
                              hipStream_t stream) {
    (void)in_sizes; (void)n_in; (void)out_size; (void)ws_size;
    const float* src   = (const float*)d_in[0];
    const float* qkv_w = (const float*)d_in[1];
    const float* qkv_b = (const float*)d_in[2];
    const float* out_w = (const float*)d_in[3];
    const float* out_b = (const float*)d_in[4];
    const float* ff1_w = (const float*)d_in[5];
    const float* ff1_b = (const float*)d_in[6];
    const float* ff2_w = (const float*)d_in[7];
    const float* ff2_b = (const float*)d_in[8];
    const float* ln1_s = (const float*)d_in[9];
    const float* ln1_b = (const float*)d_in[10];
    const float* ln2_s = (const float*)d_in[11];
    const float* ln2_b = (const float*)d_in[12];
    float* out = (float*)d_out;

    char* ws = (char*)d_ws;
    size_t off = 0;
    auto alloc = [&](size_t bytes) -> void* {
        void* p = ws + off;
        off = (off + bytes + 255) & ~(size_t)255;
        return p;
    };
    const size_t nQKVW = (size_t)NL * 3 * D * D;
    const size_t nOUTW = (size_t)NL * D * D;
    const size_t nFF1W = (size_t)NL * FF * D;
    const size_t nFF2W = (size_t)NL * D * FF;
    __bf16* Wqkv = (__bf16*)alloc(nQKVW * 2);
    __bf16* Wout = (__bf16*)alloc(nOUTW * 2);
    __bf16* Wff1 = (__bf16*)alloc(nFF1W * 2);
    __bf16* Wff2 = (__bf16*)alloc(nFF2W * 2);
    __bf16* xb   = (__bf16*)alloc((size_t)M_TOT * D * 2);
    __bf16* qkvb = (__bf16*)alloc((size_t)M_TOT * 3 * D * 2);
    __bf16* ctxb = (__bf16*)alloc((size_t)M_TOT * D * 2);
    float*  ybuf = (float*)alloc((size_t)M_TOT * D * 4);
    __bf16* hb   = (__bf16*)alloc((size_t)CHT * 128 * FF * 2);

    auto cvt = [&](const float* in, __bf16* o, size_t n) {
        k_cvt<<<(unsigned)((n + 1023) / 1024), 256, 0, stream>>>(in, o, (int)n);
    };
    cvt(qkv_w, Wqkv, nQKVW);
    cvt(out_w, Wout, nOUTW);
    cvt(ff1_w, Wff1, nFF1W);
    cvt(ff2_w, Wff2, nFF2W);
    cvt(src, xb, (size_t)M_TOT * D);

    for (int i = 0; i < NL; ++i) {
        // fused QKV projection -> bf16 [M, 768]
        k_gemm<768, 256, false, false, true><<<dim3(768 / 128, MT_TOT), 256, 0, stream>>>(
            xb, Wqkv + (size_t)i * 3 * D * D, qkv_b + (size_t)i * 3 * D, nullptr, qkvb);
        // masked attention -> bf16 ctx [M, 256]
        k_attn<<<BATCH * H, 160, 0, stream>>>(qkvb, ctxb);
        // output projection -> f32 y
        k_gemm<256, 256, false, true, false><<<dim3(D / 128, MT_TOT), 256, 0, stream>>>(
            ctxb, Wout + (size_t)i * D * D, out_b + (size_t)i * D, ybuf, nullptr);
        // x = LN(x + attn_out); writes f32 x (d_out) + bf16 x
        k_ln<<<M_TOT, 256, 0, stream>>>(
            (i == 0) ? src : out, ybuf, ln1_s + (size_t)i * D, ln1_b + (size_t)i * D,
            out, xb);
        // feed-forward, chunked over M to bound the bf16 hidden buffer
        for (int t0 = 0; t0 < MT_TOT;) {
            int nt = MT_TOT - t0; if (nt > CHT) nt = CHT;
            k_gemm<FF, 256, true, false, true><<<dim3(FF / 128, nt), 256, 0, stream>>>(
                xb + (size_t)t0 * 128 * D, Wff1 + (size_t)i * FF * D,
                ff1_b + (size_t)i * FF, nullptr, hb);
            k_gemm<256, 2048, false, true, false><<<dim3(D / 128, nt), 256, 0, stream>>>(
                hb, Wff2 + (size_t)i * D * FF, ff2_b + (size_t)i * D,
                ybuf + (size_t)t0 * 128 * D, nullptr);
            t0 += nt;
        }
        // x = LN(x + ff_out)
        k_ln<<<M_TOT, 256, 0, stream>>>(
            out, ybuf, ln2_s + (size_t)i * D, ln2_b + (size_t)i * D, out, xb);
    }
}
